// MultiHeadSelfAttention_43542378446906
// MI455X (gfx1250) — matrix operand order repaired and checked by reading, not compiled, not run on hardware
//
#include <hip/hip_runtime.h>
#include <hip/hip_bf16.h>
#include <math.h>
#include <stdint.h>

typedef __attribute__((ext_vector_type(16))) __bf16         v16bf;
typedef __attribute__((ext_vector_type(8)))  float          v8f;
typedef __attribute__((ext_vector_type(8)))  unsigned short v8us;
typedef __attribute__((ext_vector_type(4)))  int            v4i;
typedef unsigned short us;

// Problem constants
constexpr int Dm  = 1024;   // model dim
constexpr int Hh  = 16;     // heads
constexpr int DK  = 64;     // head dim
constexpr int Sq  = 2048;   // sequence
constexpr int Bb  = 2;      // batch
constexpr int Mr  = Bb * Sq;            // 4096 rows
constexpr int NX  = Mr * Dm;            // x elements
constexpr int NW  = Dm * Dm;            // weight elements
constexpr int NR  = Bb * Sq * (Dm / 2); // rope work items

__device__ __forceinline__ us f2bf(float f) {
  union { float f; unsigned u; } x; x.f = f;
  unsigned r = x.u + 0x7FFFu + ((x.u >> 16) & 1u);
  return (us)(r >> 16);
}
__device__ __forceinline__ float bf2f(us h) {
  union { unsigned u; float f; } x; x.u = ((unsigned)h) << 16;
  return x.f;
}

union Frag {           // one WMMA 16-bit A/B operand (8 VGPRs / lane)
  v8us  h[2];
  v16bf bf;
};

// ---- CDNA5 async global->LDS staging (ASYNCcnt path), with ds_store fallback ----
__device__ __forceinline__ void stage16B(const us* gsrc, us* ldst) {
#if __has_builtin(__builtin_amdgcn_global_load_async_to_lds_b128)
  using gv4i = __attribute__((address_space(1))) v4i;  // global src (64-bit ptr)
  using lv4i = __attribute__((address_space(3))) v4i;  // LDS dst (32-bit ptr)
  __builtin_amdgcn_global_load_async_to_lds_b128(
      (gv4i*)(uintptr_t)gsrc,
      (lv4i*)(unsigned)(uintptr_t)ldst, 0, 0);
#else
  *(v8us*)ldst = *(const v8us*)gsrc;
#endif
}
__device__ __forceinline__ void stage_wait() {
#if __has_builtin(__builtin_amdgcn_global_load_async_to_lds_b128)
#if __has_builtin(__builtin_amdgcn_s_wait_asynccnt)
  __builtin_amdgcn_s_wait_asynccnt(0);
#else
  asm volatile("s_wait_asynccnt 0x0" ::: "memory");
#endif
#endif
}

// ---------------- prep kernels ----------------
__global__ void k_convert(const float* __restrict__ in, us* __restrict__ out, int n) {
  int i = blockIdx.x * blockDim.x + threadIdx.x;
  if (i < n) out[i] = f2bf(in[i]);
}

// wT[d][e] = w[e][d]  (bf16)
__global__ void k_transpose(const float* __restrict__ w, us* __restrict__ wT) {
  int i = blockIdx.x * blockDim.x + threadIdx.x;
  if (i < NW) {
    int e = i >> 10, d = i & 1023;
    wT[(size_t)d * Dm + e] = f2bf(w[i]);
  }
}

// ---------------- GEMM: out[m][e] = sum_d A[m][d] * W[e][d]  (BT = W^T) ----------------
// Block = 8 waves; each wave owns a 32(M) x 64(N) tile -> block tile 256 x 64.
__global__ __launch_bounds__(256) void k_gemm(const us* __restrict__ A,
                                              const us* __restrict__ BT,
                                              us* __restrict__ outB,
                                              float* __restrict__ outF,
                                              const float* __restrict__ bias) {
  __shared__ us bs[32 * 64];
  const int tid  = threadIdx.x;
  const int lane = tid & 31;
  const int wave = tid >> 5;
  const int n0   = blockIdx.x * 64;
  const int m0   = blockIdx.y * 256 + wave * 32;
  const int ln   = lane & 15;
  const int hk   = lane >> 4;
  const size_t arow0 = (size_t)(m0 + ln) * Dm;
  const size_t arow1 = (size_t)(m0 + 16 + ln) * Dm;

  v8f acc[2][4];
#pragma unroll
  for (int t = 0; t < 2; ++t)
#pragma unroll
    for (int f = 0; f < 4; ++f)
#pragma unroll
      for (int v = 0; v < 8; ++v) acc[t][f][v] = 0.f;

  const int srow = tid >> 3, sch = tid & 7;  // staging: 256 thr x 16B = 32x64 tile
  for (int kk = 0; kk < Dm; kk += 32) {
    __syncthreads();
    stage16B(&BT[(size_t)(kk + srow) * Dm + n0 + sch * 8], &bs[srow * 64 + sch * 8]);
    stage_wait();
    __syncthreads();

    Frag a0, a1;
    a0.h[0] = *(const v8us*)&A[arow0 + kk + hk * 8];
    a0.h[1] = *(const v8us*)&A[arow0 + kk + 16 + hk * 8];
    a1.h[0] = *(const v8us*)&A[arow1 + kk + hk * 8];
    a1.h[1] = *(const v8us*)&A[arow1 + kk + 16 + hk * 8];
    if (kk + 32 < Dm) {
      __builtin_prefetch(&A[arow0 + kk + 32], 0, 0);
      __builtin_prefetch(&A[arow1 + kk + 32], 0, 0);
    }
    Frag bfr[4];  // preload ALL B fragments, then issue WMMAs back-to-back
#pragma unroll
    for (int f = 0; f < 4; ++f) {
#pragma unroll
      for (int s = 0; s < 8; ++s) {
        bfr[f].h[0][s] = bs[(hk * 8 + s) * 64 + f * 16 + ln];
        bfr[f].h[1][s] = bs[(16 + hk * 8 + s) * 64 + f * 16 + ln];
      }
    }
#pragma unroll
    for (int f = 0; f < 4; ++f) {
      acc[0][f] = __builtin_amdgcn_wmma_f32_16x16x32_bf16(false, a0.bf, false, bfr[f].bf,
                                                          (short)0, acc[0][f], false, false);
      acc[1][f] = __builtin_amdgcn_wmma_f32_16x16x32_bf16(false, a1.bf, false, bfr[f].bf,
                                                          (short)0, acc[1][f], false, false);
    }
  }
#pragma unroll
  for (int t = 0; t < 2; ++t)
#pragma unroll
    for (int f = 0; f < 4; ++f)
#pragma unroll
      for (int v = 0; v < 8; ++v) {
        int row = m0 + t * 16 + hk * 8 + v;
        int col = n0 + f * 16 + ln;
        float val = acc[t][f][v];
        if (outF) outF[(size_t)row * Dm + col] = val + bias[col];
        else      outB[(size_t)row * Dm + col] = f2bf(val);
      }
}

// ---------------- RoPE (split-halves) + K transpose ----------------
__global__ void k_rope(const us* __restrict__ Qraw, const us* __restrict__ Kraw,
                       us* __restrict__ Qr, us* __restrict__ KT) {
  int i = blockIdx.x * blockDim.x + threadIdx.x;
  if (i >= NR) return;
  int j = i & 511;            // pair index 0..511
  int s = (i >> 9) & 2047;
  int b = i >> 20;            // 512*2048 = 2^20
  size_t base = (size_t)(b * Sq + s) * Dm;
  float freq = __expf((float)(2 * j) * (-9.210340371976184f / 1024.f)); // exp(-2j*ln(1e4)/D)
  float ang = (float)s * freq;
  float sn, c; __sincosf(ang, &sn, &c);
  float qe = bf2f(Qraw[base + 2 * j]), qo = bf2f(Qraw[base + 2 * j + 1]);
  float ke = bf2f(Kraw[base + 2 * j]), ko = bf2f(Kraw[base + 2 * j + 1]);
  Qr[base + j]       = f2bf(qe * c - qo * sn);
  Qr[base + 512 + j] = f2bf(qe * sn + qo * c);
  // KT[b*1024 + e][s]
  KT[(size_t)(b * Dm + j) * Sq + s]       = f2bf(ke * c - ko * sn);
  KT[(size_t)(b * Dm + 512 + j) * Sq + s] = f2bf(ke * sn + ko * c);
}

// ---------------- Flash attention (online softmax, bf16 WMMA, f32 state) ----------------
__global__ __launch_bounds__(256) void k_attn(const us* __restrict__ Q,
                                              const us* __restrict__ KT,
                                              const us* __restrict__ V,
                                              us* __restrict__ O) {
  __shared__ us pb[8 * 16 * 32];  // per-wave 16x32 P tile
  const int lane = threadIdx.x & 31;
  const int wave = threadIdx.x >> 5;
  const int bh = blockIdx.y, b = bh >> 4, h = bh & 15;
  const int q0 = blockIdx.x * 128 + wave * 16;
  const int ln = lane & 15, hk = lane >> 4;
  us* pw = pb + wave * 512;

  // Q A-fragments (16 x 64, two K-chunks of 32)
  const us* qrow = Q + (size_t)(b * Sq + q0 + ln) * Dm + h * DK;
  Frag qa0, qa1;
  qa0.h[0] = *(const v8us*)(qrow + hk * 8);
  qa0.h[1] = *(const v8us*)(qrow + 16 + hk * 8);
  qa1.h[0] = *(const v8us*)(qrow + 32 + hk * 8);
  qa1.h[1] = *(const v8us*)(qrow + 48 + hk * 8);

  float rm[8], rl[8], fac[8];
  v8f o[4];
#pragma unroll
  for (int v = 0; v < 8; ++v) { rm[v] = -1e30f; rl[v] = 0.f; }
#pragma unroll
  for (int f = 0; f < 4; ++f)
#pragma unroll
    for (int v = 0; v < 8; ++v) o[f][v] = 0.f;

  const us* kbase = KT + (size_t)(b * Dm + h * DK) * Sq;
  const us* vbase = V + (size_t)(b * Sq) * Dm + h * DK;

  for (int key0 = 0; key0 < Sq; key0 += 32) {
    // K^T B-fragments: lane l <-> key column l%16, elements over dk (table order; chunk1 = dk+32)
    const us* kr0 = kbase + key0 + ln;
    const us* kr1 = kbase + (size_t)32 * Sq + key0 + ln;
    Frag kb00, kb01, kb10, kb11;
#pragma unroll
    for (int s = 0; s < 8; ++s) {
      size_t d0 = (size_t)(hk * 8 + s) * Sq, d1 = (size_t)(16 + hk * 8 + s) * Sq;
      kb00.h[0][s] = kr0[d0];      kb00.h[1][s] = kr0[d1];
      kb01.h[0][s] = kr0[d0 + 16]; kb01.h[1][s] = kr0[d1 + 16];
      kb10.h[0][s] = kr1[d0];      kb10.h[1][s] = kr1[d1];
      kb11.h[0][s] = kr1[d0 + 16]; kb11.h[1][s] = kr1[d1 + 16];
    }

    v8f s0, s1;
#pragma unroll
    for (int v = 0; v < 8; ++v) { s0[v] = 0.f; s1[v] = 0.f; }
    s0 = __builtin_amdgcn_wmma_f32_16x16x32_bf16(false, qa0.bf, false, kb00.bf, (short)0, s0, false, false);
    s0 = __builtin_amdgcn_wmma_f32_16x16x32_bf16(false, qa1.bf, false, kb10.bf, (short)0, s0, false, false);
    s1 = __builtin_amdgcn_wmma_f32_16x16x32_bf16(false, qa0.bf, false, kb01.bf, (short)0, s1, false, false);
    s1 = __builtin_amdgcn_wmma_f32_16x16x32_bf16(false, qa1.bf, false, kb11.bf, (short)0, s1, false, false);

    float mb[8], ls[8];
#pragma unroll
    for (int v = 0; v < 8; ++v) {
      float a = s0[v] * 0.125f, c = s1[v] * 0.125f;  // 1/sqrt(64)
      s0[v] = a; s1[v] = c;
      mb[v] = fmaxf(a, c);
    }
#pragma unroll
    for (int v = 0; v < 8; ++v) {  // half-wave (row) max
      mb[v] = fmaxf(mb[v], __shfl_xor(mb[v], 1, 32));
      mb[v] = fmaxf(mb[v], __shfl_xor(mb[v], 2, 32));
      mb[v] = fmaxf(mb[v], __shfl_xor(mb[v], 4, 32));
      mb[v] = fmaxf(mb[v], __shfl_xor(mb[v], 8, 32));
    }
#pragma unroll
    for (int v = 0; v < 8; ++v) {
      float mn = fmaxf(rm[v], mb[v]);
      fac[v] = __expf(rm[v] - mn);
      rm[v] = mn;
    }
#pragma unroll
    for (int v = 0; v < 8; ++v) {
      float p0 = __expf(s0[v] - rm[v]);
      float p1 = __expf(s1[v] - rm[v]);
      int r = hk * 8 + v;
      pw[r * 32 + ln]      = f2bf(p0);
      pw[r * 32 + 16 + ln] = f2bf(p1);
      ls[v] = p0 + p1;
    }
#pragma unroll
    for (int v = 0; v < 8; ++v) {  // half-wave row sum
      ls[v] += __shfl_xor(ls[v], 1, 32);
      ls[v] += __shfl_xor(ls[v], 2, 32);
      ls[v] += __shfl_xor(ls[v], 4, 32);
      ls[v] += __shfl_xor(ls[v], 8, 32);
    }
#pragma unroll
    for (int v = 0; v < 8; ++v) rl[v] = rl[v] * fac[v] + ls[v];
#pragma unroll
    for (int f = 0; f < 4; ++f)
#pragma unroll
      for (int v = 0; v < 8; ++v) o[f][v] *= fac[v];

    asm volatile("s_wait_dscnt 0" ::: "memory");  // P stores -> A-frag loads (same wave)

    Frag pa;  // re-read P in A-operand layout
    pa.h[0] = *(const v8us*)&pw[ln * 32 + hk * 8];
    pa.h[1] = *(const v8us*)&pw[ln * 32 + 16 + hk * 8];

    const us* vr = vbase + (size_t)key0 * Dm + ln;  // lane l <-> dk column l%16, elements over key
    Frag vb[4];  // preload all V fragments, then issue WMMAs back-to-back
#pragma unroll
    for (int f = 0; f < 4; ++f) {
#pragma unroll
      for (int s = 0; s < 8; ++s) {
        vb[f].h[0][s] = vr[(size_t)(hk * 8 + s) * Dm + f * 16];
        vb[f].h[1][s] = vr[(size_t)(16 + hk * 8 + s) * Dm + f * 16];
      }
    }
#pragma unroll
    for (int f = 0; f < 4; ++f)
      o[f] = __builtin_amdgcn_wmma_f32_16x16x32_bf16(false, pa.bf, false, vb[f].bf,
                                                     (short)0, o[f], false, false);
  }

#pragma unroll
  for (int f = 0; f < 4; ++f)
#pragma unroll
    for (int v = 0; v < 8; ++v) {
      int row = q0 + hk * 8 + v;
      int col = h * DK + f * 16 + ln;
      O[(size_t)(b * Sq + row) * Dm + col] = f2bf(o[f][v] / rl[v]);
    }
}

// ---------------- host ----------------
extern "C" void kernel_launch(void* const* d_in, const int* in_sizes, int n_in,
                              void* d_out, int out_size, void* d_ws, size_t ws_size,
                              hipStream_t stream) {
  (void)in_sizes; (void)n_in; (void)out_size; (void)ws_size;
  const float* x   = (const float*)d_in[0];
  const float* wq  = (const float*)d_in[1];
  const float* wk  = (const float*)d_in[2];
  const float* wv  = (const float*)d_in[3];
  const float* wo  = (const float*)d_in[4];
  const float* wob = (const float*)d_in[5];
  float* out = (float*)d_out;

  char* ws = (char*)d_ws;
  const size_t MB = 1u << 20;
  us* xb    = (us*)(ws);                 //  8 MB  x bf16
  us* wqT   = (us*)(ws + 8  * MB);       //  2 MB
  us* wkT   = (us*)(ws + 10 * MB);       //  2 MB
  us* wvT   = (us*)(ws + 12 * MB);       //  2 MB
  us* woT   = (us*)(ws + 14 * MB);       //  2 MB
  us* Qraw  = (us*)(ws + 16 * MB);       //  8 MB
  us* Kraw  = (us*)(ws + 24 * MB);       //  8 MB
  us* Vraw  = (us*)(ws + 32 * MB);       //  8 MB
  us* Qr    = (us*)(ws + 40 * MB);       //  8 MB
  us* KTb   = (us*)(ws + 48 * MB);       //  8 MB
  us* attnb = (us*)(ws + 56 * MB);       //  8 MB  (total 64 MB)

  k_convert<<<(NX + 255) / 256, 256, 0, stream>>>(x, xb, NX);
  k_transpose<<<(NW + 255) / 256, 256, 0, stream>>>(wq, wqT);
  k_transpose<<<(NW + 255) / 256, 256, 0, stream>>>(wk, wkT);
  k_transpose<<<(NW + 255) / 256, 256, 0, stream>>>(wv, wvT);
  k_transpose<<<(NW + 255) / 256, 256, 0, stream>>>(wo, woT);

  dim3 ggrid(Dm / 64, Mr / 256);  // (16, 16)
  k_gemm<<<ggrid, 256, 0, stream>>>(xb, wqT, Qraw, nullptr, nullptr);
  k_gemm<<<ggrid, 256, 0, stream>>>(xb, wkT, Kraw, nullptr, nullptr);
  k_gemm<<<ggrid, 256, 0, stream>>>(xb, wvT, Vraw, nullptr, nullptr);

  k_rope<<<(NR + 255) / 256, 256, 0, stream>>>(Qraw, Kraw, Qr, KTb);

  dim3 agrid(Sq / 128, Bb * Hh);  // (16, 32)
  k_attn<<<agrid, 256, 0, stream>>>(Qr, KTb, Vraw, attnb);

  k_gemm<<<ggrid, 256, 0, stream>>>(attnb, woT, nullptr, out, wob);
}
